// RetNetBlock_70566312673631
// MI455X (gfx1250) — matrix operand order repaired and checked by reading, not compiled, not run on hardware
//
#include <hip/hip_runtime.h>
#include <hip/hip_bf16.h>

// ---------------------------------------------------------------------------
// RetNet block for MI455X (gfx1250, wave32, WMMA).
// GEMMs: v_wmma_f32_16x16x32_bf16, fp32 accumulate, transposed-B LDS layout
// so every fragment build is contiguous ds_load_b128. Global->LDS staging
// uses async-to-LDS DMA (ASYNCcnt) when the builtin is available.
// ---------------------------------------------------------------------------

typedef __attribute__((ext_vector_type(16))) __bf16 bf16x16;
typedef __attribute__((ext_vector_type(8)))  float  f32x8;
typedef __attribute__((ext_vector_type(4)))  int    v4i;

#define NB_B      2
#define NB_T      2048
#define NB_D      2048
#define NB_H      8
#define NB_DK     256
#define NB_DV     512
#define NB_INTER  5632
#define NB_BT     (NB_B * NB_T)
#define NB_CHUNK  64

#if __has_builtin(__builtin_amdgcn_global_load_async_to_lds_b128)
#define NB_ASYNC 1
#endif

#ifdef NB_ASYNC
__device__ __forceinline__ void async_cp_b128(const void* g, void* l) {
    __builtin_amdgcn_global_load_async_to_lds_b128(
        (__attribute__((address_space(1))) v4i*)(uintptr_t)g,
        (__attribute__((address_space(3))) v4i*)(uint32_t)(uintptr_t)l,
        0, 0);
}
__device__ __forceinline__ void wait_async() {
#if __has_builtin(__builtin_amdgcn_s_wait_asynccnt)
    __builtin_amdgcn_s_wait_asynccnt(0);
#else
    asm volatile("s_wait_asynccnt 0" ::: "memory");
#endif
}
#else
__device__ __forceinline__ void wait_async() {}
#endif

// ISA 16-bit A-matrix 16x32 layout: lane<16 -> K {0..7,16..23}, lane>=16 -> K {8..15,24..31}
__device__ __forceinline__ int wmma_a_k(int e, int hf) {
    int v = e >> 1, p = e & 1;
    return (v < 4) ? (hf * 8 + v * 2 + p) : (16 + hf * 8 + (v - 4) * 2 + p);
}

__device__ __forceinline__ unsigned pack2_bf16(float a, float b) {
    unsigned short s0 = __builtin_bit_cast(unsigned short, (__bf16)a);
    unsigned short s1 = __builtin_bit_cast(unsigned short, (__bf16)b);
    return (unsigned)s0 | ((unsigned)s1 << 16);
}

// ---------------------------------------------------------------------------
// GEMM: C[M,N] = A[M,K] * Bt[N,K]^T (+R). Block 128x128, 8 waves of 64x32,
// K step 32, double-buffered async LDS staging.
// mode 0: fp32 out; mode 1: bf16 out; mode 2: fp32 out + residual add.
// ---------------------------------------------------------------------------
__global__ __launch_bounds__(256)
void gemm_bf16_wmma(const __bf16* __restrict__ A, const __bf16* __restrict__ Bt,
                    const float* __restrict__ R, void* __restrict__ Cout,
                    int M, int N, int K, int mode)
{
    __shared__ __bf16 As[2][128][40];
    __shared__ __bf16 Bs[2][128][40];

    const int tid  = threadIdx.x;
    const int lane = tid & 31;
    const int wave = tid >> 5;
    const int wm   = wave >> 2;      // 0..1
    const int wn   = wave & 3;       // 0..3
    const int hf   = lane >> 4;
    const int mn   = lane & 15;
    const int bm   = blockIdx.y * 128;
    const int bn   = blockIdx.x * 128;

    const int sr = tid >> 2;          // 0..63
    const int sc = (tid & 3) * 8;     // 0,8,16,24

    f32x8 acc[4][2] = {};

    auto stage = [&](int buf, int kk) {
        const __bf16* ga0 = &A [(size_t)(bm + sr)      * K + kk + sc];
        const __bf16* ga1 = &A [(size_t)(bm + sr + 64) * K + kk + sc];
        const __bf16* gb0 = &Bt[(size_t)(bn + sr)      * K + kk + sc];
        const __bf16* gb1 = &Bt[(size_t)(bn + sr + 64) * K + kk + sc];
#ifdef NB_ASYNC
        async_cp_b128(ga0, &As[buf][sr][sc]);
        async_cp_b128(ga1, &As[buf][sr + 64][sc]);
        async_cp_b128(gb0, &Bs[buf][sr][sc]);
        async_cp_b128(gb1, &Bs[buf][sr + 64][sc]);
#else
        if (kk + 32 < K) {
            __builtin_prefetch(ga0 + 32, 0, 0);
            __builtin_prefetch(gb0 + 32, 0, 0);
        }
        *(uint4*)&As[buf][sr][sc]      = *(const uint4*)ga0;
        *(uint4*)&As[buf][sr + 64][sc] = *(const uint4*)ga1;
        *(uint4*)&Bs[buf][sr][sc]      = *(const uint4*)gb0;
        *(uint4*)&Bs[buf][sr + 64][sc] = *(const uint4*)gb1;
#endif
    };

    stage(0, 0);
    int cur = 0;
    for (int kk = 0; kk < K; kk += 32) {
        wait_async();
        __syncthreads();
        if (kk + 32 < K) stage(cur ^ 1, kk + 32);

        bf16x16 bfrag[2];
        #pragma unroll
        for (int tj = 0; tj < 2; ++tj) {
            #pragma unroll
            for (int e = 0; e < 16; ++e)
                bfrag[tj][e] = Bs[cur][wn * 32 + tj * 16 + mn][hf * 16 + e];
        }
        #pragma unroll
        for (int ti = 0; ti < 4; ++ti) {
            bf16x16 afrag;
            #pragma unroll
            for (int e = 0; e < 16; ++e)
                afrag[e] = As[cur][wm * 64 + ti * 16 + mn][hf * 16 + e];
            #pragma unroll
            for (int tj = 0; tj < 2; ++tj)
                acc[ti][tj] = __builtin_amdgcn_wmma_f32_16x16x32_bf16(
                    false, afrag, false, bfrag[tj], (short)0, acc[ti][tj], false, false);
        }
        __syncthreads();
        cur ^= 1;
    }

    #pragma unroll
    for (int ti = 0; ti < 4; ++ti) {
        #pragma unroll
        for (int tj = 0; tj < 2; ++tj) {
            int row0 = bm + wm * 64 + ti * 16 + hf * 8;
            int col  = bn + wn * 32 + tj * 16 + mn;
            #pragma unroll
            for (int r = 0; r < 8; ++r) {
                size_t idx = (size_t)(row0 + r) * N + col;
                float v = acc[ti][tj][r];
                if (mode == 2) v += R[idx];
                if (mode == 1) ((__bf16*)Cout)[idx] = (__bf16)v;
                else           ((float*)Cout)[idx]  = v;
            }
        }
    }
}

// ---------------------------------------------------------------------------
// Transposing fp32 -> bf16 weight cast: dst[N][K] = (bf16)src[K][N]^T
// ---------------------------------------------------------------------------
__global__ __launch_bounds__(256)
void transpose_cast_bf16(const float* __restrict__ src, __bf16* __restrict__ dst,
                         int K, int N)
{
    __shared__ __bf16 t[32][40];
    int k0 = blockIdx.y * 32, n0 = blockIdx.x * 32;
    int r = threadIdx.x >> 3, cq = (threadIdx.x & 7) * 4;
    float4 f = *(const float4*)&src[(size_t)(k0 + r) * N + n0 + cq];
    t[r][cq + 0] = (__bf16)f.x; t[r][cq + 1] = (__bf16)f.y;
    t[r][cq + 2] = (__bf16)f.z; t[r][cq + 3] = (__bf16)f.w;
    __syncthreads();
    unsigned short s0 = __builtin_bit_cast(unsigned short, t[cq + 0][r]);
    unsigned short s1 = __builtin_bit_cast(unsigned short, t[cq + 1][r]);
    unsigned short s2 = __builtin_bit_cast(unsigned short, t[cq + 2][r]);
    unsigned short s3 = __builtin_bit_cast(unsigned short, t[cq + 3][r]);
    uint2 o2 = make_uint2((unsigned)s0 | ((unsigned)s1 << 16),
                          (unsigned)s2 | ((unsigned)s3 << 16));
    *(uint2*)&dst[(size_t)(n0 + r) * K + k0 + cq] = o2;
}

// ---------------------------------------------------------------------------
// RMSNorm (fp32 in, bf16 out), one row per block.
// ---------------------------------------------------------------------------
__global__ __launch_bounds__(256)
void rmsnorm_bf16_kernel(const float* __restrict__ x, const float* __restrict__ w,
                         __bf16* __restrict__ out, int d)
{
    int row = blockIdx.x;
    const float* xr = x + (size_t)row * d;
    __shared__ float red[8];
    float ss = 0.f;
    for (int i = threadIdx.x; i < d; i += 256) { float v = xr[i]; ss += v * v; }
    #pragma unroll
    for (int off = 16; off > 0; off >>= 1) ss += __shfl_down(ss, off, 32);
    if ((threadIdx.x & 31) == 0) red[threadIdx.x >> 5] = ss;
    __syncthreads();
    if (threadIdx.x == 0) {
        float t = 0.f;
        #pragma unroll
        for (int i = 0; i < 8; ++i) t += red[i];
        red[0] = rsqrtf(t / (float)d + 1e-6f);
    }
    __syncthreads();
    float sc = red[0];
    for (int i = threadIdx.x; i < d; i += 256)
        out[(size_t)row * d + i] = (__bf16)(xr[i] * sc * w[i]);
}

// ---------------------------------------------------------------------------
// RoPE (non-interleaved, head dim 256) on q and k; q scaled by DK^-0.5.
// grid (BT, H), block 128.
// ---------------------------------------------------------------------------
__global__ __launch_bounds__(128)
void rope_scale_kernel(float* __restrict__ q, float* __restrict__ k)
{
    int tok = blockIdx.x, h = blockIdx.y;
    int t = tok % NB_T;
    int p = threadIdx.x;
    size_t base = (size_t)tok * (NB_H * NB_DK) + (size_t)h * NB_DK;
    float freq = exp2f(-(2.0f * (float)p / 256.0f) * 13.28771238f);  // log2(1e4)
    float ang = (float)t * freq;
    float s, c;
    sincosf(ang, &s, &c);
    float q1 = q[base + p], q2 = q[base + 128 + p];
    q[base + p]       = (q1 * c - q2 * s) * 0.0625f;
    q[base + 128 + p] = (q2 * c + q1 * s) * 0.0625f;
    float k1 = k[base + p], k2 = k[base + 128 + p];
    k[base + p]       = k1 * c - k2 * s;
    k[base + 128 + p] = k2 * c + k1 * s;
}

// ---------------------------------------------------------------------------
// Chunkwise retention. grid = B*H*8; each block owns a 64-wide Dv slice and
// keeps 256x64 fp32 state (+ bf16 transposed mirror) in LDS over 32 chunks.
// All operand layouts are K-contiguous so fragment builds are ds_load_b128.
// LDS map (bytes):
//   S    f32 [256][64]   65536 @      0
//   qs  bf16 [64][264]   33792 @  65536
//   ks  bf16 [64][264]   33792 @  99328
//   kst bf16 [256][72]   36864 @ 133120   (k^T, pre-scaled by k_decay)
//   vst bf16 [64][72]     9216 @ 169984   (v^T)
//   Sbt bf16 [64][264]   33792 @ 179200   (S^T, bf16 mirror)
//   att bf16 [64][72]     9216 @ 212992
//   kdec f32 [64]          256 @ 222208
// ---------------------------------------------------------------------------
#define RET_SMEM 222464
__global__ __launch_bounds__(256)
void retention_kernel(const float* __restrict__ q, const float* __restrict__ k,
                      const float* __restrict__ v, float* __restrict__ o)
{
    extern __shared__ char smem[];
    float  (*S)[64]     = (float (*)[64])   (smem);
    __bf16 (*qs)[264]   = (__bf16 (*)[264]) (smem + 65536);
    __bf16 (*ks)[264]   = (__bf16 (*)[264]) (smem + 99328);
    __bf16 (*kst)[72]   = (__bf16 (*)[72])  (smem + 133120);
    __bf16 (*vst)[72]   = (__bf16 (*)[72])  (smem + 169984);
    __bf16 (*Sbt)[264]  = (__bf16 (*)[264]) (smem + 179200);
    __bf16 (*attnb)[72] = (__bf16 (*)[72])  (smem + 212992);
    float*  kdec        = (float*)          (smem + 222208);

    const int bid  = blockIdx.x;
    const int b    = bid / (NB_H * 8);
    const int h    = (bid / 8) % NB_H;
    const int dsl  = bid & 7;                 // Dv slice of 64
    const int tid  = threadIdx.x;
    const int lane = tid & 31;
    const int wave = tid >> 5;
    const int hf   = lane >> 4;
    const int mn   = lane & 15;

    const float gamma = 1.0f - exp2f(-5.0f - (float)h);
    const float lg = log2f(gamma);
    const float chunk_decay = exp2f(64.0f * lg);
    const int ldqk = NB_H * NB_DK;            // 2048
    const int ldv  = NB_H * NB_DV;            // 4096

    for (int i = tid; i < 16384; i += 256) ((float*)smem)[i] = 0.f;          // S
    for (int i = tid; i < 16896; i += 256) ((unsigned short*)(smem + 179200))[i] = 0; // Sbt
    if (tid < 64) kdec[tid] = exp2f((float)(63 - tid) * lg);
    __syncthreads();

    for (int n = 0; n < NB_T / NB_CHUNK; ++n) {
        const int row0 = b * NB_T + n * NB_CHUNK;
        // ---- stage chunk: q,k (row + scaled-transposed k), v (transposed) ----
        for (int g = tid; g < 4096; g += 256) {            // 64 x 256
            int r = g >> 6, c = (g & 63) * 4;
            float4 fq = *(const float4*)&q[(size_t)(row0 + r) * ldqk + h * NB_DK + c];
            float4 fk = *(const float4*)&k[(size_t)(row0 + r) * ldqk + h * NB_DK + c];
            *(uint2*)&qs[r][c] = make_uint2(pack2_bf16(fq.x, fq.y), pack2_bf16(fq.z, fq.w));
            *(uint2*)&ks[r][c] = make_uint2(pack2_bf16(fk.x, fk.y), pack2_bf16(fk.z, fk.w));
            float kd = kdec[r];
            kst[c + 0][r] = (__bf16)(fk.x * kd); kst[c + 1][r] = (__bf16)(fk.y * kd);
            kst[c + 2][r] = (__bf16)(fk.z * kd); kst[c + 3][r] = (__bf16)(fk.w * kd);
        }
        for (int g = tid; g < 1024; g += 256) {            // 64 x 64 slice
            int r = g >> 4, c = (g & 15) * 4;
            float4 fv = *(const float4*)&v[(size_t)(row0 + r) * ldv + h * NB_DV + dsl * 64 + c];
            vst[c + 0][r] = (__bf16)fv.x; vst[c + 1][r] = (__bf16)fv.y;
            vst[c + 2][r] = (__bf16)fv.z; vst[c + 3][r] = (__bf16)fv.w;
        }
        __syncthreads();

        // ---- stage 2: attn = (qb kb^T) .* Dmask   (16 subtiles, 2/wave) ----
        for (int s = wave; s < 16; s += 8) {
            int ti = s >> 2, tj = s & 3;
            f32x8 acc = {};
            for (int kk = 0; kk < 256; kk += 32) {
                bf16x16 afrag, bfrag;
                #pragma unroll
                for (int e = 0; e < 16; ++e) {
                    afrag[e] = qs[ti * 16 + mn][kk + hf * 16 + e];
                    bfrag[e] = ks[tj * 16 + mn][kk + hf * 16 + e];
                }
                acc = __builtin_amdgcn_wmma_f32_16x16x32_bf16(
                    false, afrag, false, bfrag, (short)0, acc, false, false);
            }
            #pragma unroll
            for (int r = 0; r < 8; ++r) {
                int i = ti * 16 + hf * 8 + r, j = tj * 16 + mn;
                float val = (i >= j) ? acc[r] * exp2f((float)(i - j) * lg) : 0.f;
                attnb[i][j] = (__bf16)val;
            }
        }
        __syncthreads();

        // ---- stage 3: o = attn @ v + gamma^(i+1) * (qb @ S)  (16 subtiles) ----
        for (int s = wave * 2; s < wave * 2 + 2; ++s) {
            int ti = s >> 2, tn = s & 3;
            f32x8 acc1 = {};
            for (int kk = 0; kk < 64; kk += 32) {
                bf16x16 afrag, bfrag;
                #pragma unroll
                for (int e = 0; e < 16; ++e) {
                    afrag[e] = attnb[ti * 16 + mn][kk + hf * 16 + e];
                    bfrag[e] = vst[tn * 16 + mn][kk + hf * 16 + e];
                }
                acc1 = __builtin_amdgcn_wmma_f32_16x16x32_bf16(
                    false, afrag, false, bfrag, (short)0, acc1, false, false);
            }
            f32x8 acc2 = {};
            for (int kk = 0; kk < 256; kk += 32) {
                bf16x16 afrag, bfrag;
                #pragma unroll
                for (int e = 0; e < 16; ++e) {
                    afrag[e] = qs[ti * 16 + mn][kk + hf * 16 + e];
                    bfrag[e] = Sbt[tn * 16 + mn][kk + hf * 16 + e];
                }
                acc2 = __builtin_amdgcn_wmma_f32_16x16x32_bf16(
                    false, afrag, false, bfrag, (short)0, acc2, false, false);
            }
            #pragma unroll
            for (int r = 0; r < 8; ++r) {
                int i = ti * 16 + hf * 8 + r;
                float cross = exp2f((float)(i + 1) * lg);
                o[(size_t)(row0 + i) * ldv + h * NB_DV + dsl * 64 + tn * 16 + mn]
                    = acc1[r] + cross * acc2[r];
            }
        }
        __syncthreads();

        // ---- stage 4: S = chunk_decay*S + (k*kdec)^T @ v  (64 subtiles) ----
        for (int s = wave * 8; s < wave * 8 + 8; ++s) {
            int td = s >> 2, tn = s & 3;
            f32x8 acc = {};
            for (int kk = 0; kk < 64; kk += 32) {
                bf16x16 afrag, bfrag;
                #pragma unroll
                for (int e = 0; e < 16; ++e) {
                    afrag[e] = kst[td * 16 + mn][kk + hf * 16 + e];
                    bfrag[e] = vst[tn * 16 + mn][kk + hf * 16 + e];
                }
                acc = __builtin_amdgcn_wmma_f32_16x16x32_bf16(
                    false, afrag, false, bfrag, (short)0, acc, false, false);
            }
            #pragma unroll
            for (int r = 0; r < 8; ++r) {
                int d = td * 16 + hf * 8 + r, e = tn * 16 + mn;
                float sv = chunk_decay * S[d][e] + acc[r];
                S[d][e] = sv;
                Sbt[e][d] = (__bf16)sv;
            }
        }
        __syncthreads();
    }
}

// ---------------------------------------------------------------------------
// Per-head RMSNorm over Dv + swish gate. grid (BT, H), block 256.
// ---------------------------------------------------------------------------
__global__ __launch_bounds__(256)
void gated_rmsnorm_kernel(const float* __restrict__ o, const float* __restrict__ g,
                          const float* __restrict__ gw, __bf16* __restrict__ out)
{
    int tok = blockIdx.x, h = blockIdx.y;
    size_t base = (size_t)tok * (NB_H * NB_DV) + (size_t)h * NB_DV;
    __shared__ float red[8];
    float ss = 0.f;
    for (int i = threadIdx.x; i < NB_DV; i += 256) { float v = o[base + i]; ss += v * v; }
    #pragma unroll
    for (int off = 16; off > 0; off >>= 1) ss += __shfl_down(ss, off, 32);
    if ((threadIdx.x & 31) == 0) red[threadIdx.x >> 5] = ss;
    __syncthreads();
    if (threadIdx.x == 0) {
        float t = 0.f;
        #pragma unroll
        for (int i = 0; i < 8; ++i) t += red[i];
        red[0] = rsqrtf(t / (float)NB_DV + 1e-6f);
    }
    __syncthreads();
    float sc = red[0];
    for (int i = threadIdx.x; i < NB_DV; i += 256) {
        float gv = g[base + i];
        float sig = 1.f / (1.f + __expf(-gv));
        out[base + i] = (__bf16)(o[base + i] * sc * gw[i] * gv * sig);
    }
}

// ---------------------------------------------------------------------------
// SwiGLU: act = silu(y[:, :INTER]) * y[:, INTER:]  (fp32 -> bf16)
// ---------------------------------------------------------------------------
__global__ __launch_bounds__(256)
void swiglu_kernel(const float* __restrict__ y, __bf16* __restrict__ act)
{
    const int n = NB_BT * NB_INTER;
    for (int idx = blockIdx.x * 256 + threadIdx.x; idx < n; idx += gridDim.x * 256) {
        int row = idx / NB_INTER, c = idx % NB_INTER;
        float gate = y[(size_t)row * (2 * NB_INTER) + c];
        float up   = y[(size_t)row * (2 * NB_INTER) + NB_INTER + c];
        float sig  = 1.f / (1.f + __expf(-gate));
        act[idx] = (__bf16)(gate * sig * up);
    }
}

// ---------------------------------------------------------------------------
// Workspace layout (bytes). ybuf aliases q/k/v/g, act_bf aliases obuf.
// ---------------------------------------------------------------------------
#define O_WQ     0ull
#define O_WK     8388608ull
#define O_WV     16777216ull
#define O_WG     33554432ull
#define O_WO     50331648ull
#define O_WGATE  67108864ull
#define O_WDOWN  113246208ull
#define O_HBF    136314880ull
#define O_QF     153092096ull
#define O_KF     186646528ull
#define O_VF     220200960ull
#define O_GF     287309824ull
#define O_OF     354418688ull
#define O_OGBF   421527552ull
#define O_RESID  455081984ull
#define O_H2BF   488636416ull
#define O_YF     O_QF
#define O_ACTBF  O_OF

extern "C" void kernel_launch(void* const* d_in, const int* in_sizes, int n_in,
                              void* d_out, int out_size, void* d_ws, size_t ws_size,
                              hipStream_t stream) {
    const float* hidden      = (const float*)d_in[0];
    const float* attn_norm_w = (const float*)d_in[1];
    const float* q_w         = (const float*)d_in[2];
    const float* k_w         = (const float*)d_in[3];
    const float* v_w         = (const float*)d_in[4];
    const float* g_w         = (const float*)d_in[5];
    const float* o_w         = (const float*)d_in[6];
    const float* gnorm_w     = (const float*)d_in[7];
    const float* mlp_norm_w  = (const float*)d_in[8];
    const float* gate_w      = (const float*)d_in[9];
    const float* down_w      = (const float*)d_in[10];

    char* ws = (char*)d_ws;
    __bf16* wq    = (__bf16*)(ws + O_WQ);
    __bf16* wk    = (__bf16*)(ws + O_WK);
    __bf16* wv    = (__bf16*)(ws + O_WV);
    __bf16* wg    = (__bf16*)(ws + O_WG);
    __bf16* wo    = (__bf16*)(ws + O_WO);
    __bf16* wgate = (__bf16*)(ws + O_WGATE);
    __bf16* wdown = (__bf16*)(ws + O_WDOWN);
    __bf16* hbf   = (__bf16*)(ws + O_HBF);
    float*  qf    = (float*) (ws + O_QF);
    float*  kf    = (float*) (ws + O_KF);
    float*  vf    = (float*) (ws + O_VF);
    float*  gf    = (float*) (ws + O_GF);
    float*  of    = (float*) (ws + O_OF);
    __bf16* ogbf  = (__bf16*)(ws + O_OGBF);
    float*  resid = (float*) (ws + O_RESID);
    __bf16* h2bf  = (__bf16*)(ws + O_H2BF);
    float*  yf    = (float*) (ws + O_YF);
    __bf16* actbf = (__bf16*)(ws + O_ACTBF);

    // transposed bf16 weights: dst[N][K]
    auto tcast = [&](const float* s, __bf16* dst, int K, int N) {
        transpose_cast_bf16<<<dim3(N / 32, K / 32), 256, 0, stream>>>(s, dst, K, N);
    };
    tcast(q_w,    wq,    NB_D,     NB_D);
    tcast(k_w,    wk,    NB_D,     NB_D);
    tcast(v_w,    wv,    NB_D,     2 * NB_D);
    tcast(g_w,    wg,    NB_D,     2 * NB_D);
    tcast(o_w,    wo,    2 * NB_D, NB_D);
    tcast(gate_w, wgate, NB_D,     2 * NB_INTER);
    tcast(down_w, wdown, NB_INTER, NB_D);

    rmsnorm_bf16_kernel<<<NB_BT, 256, 0, stream>>>(hidden, attn_norm_w, hbf, NB_D);

    auto gemm = [&](const __bf16* A, const __bf16* Bm, const float* R, void* C,
                    int M, int N, int K, int mode) {
        gemm_bf16_wmma<<<dim3(N / 128, M / 128), 256, 0, stream>>>(A, Bm, R, C, M, N, K, mode);
    };
    gemm(hbf, wq, nullptr, qf, NB_BT, NB_D,     NB_D, 0);
    gemm(hbf, wk, nullptr, kf, NB_BT, NB_D,     NB_D, 0);
    gemm(hbf, wv, nullptr, vf, NB_BT, 2 * NB_D, NB_D, 0);
    gemm(hbf, wg, nullptr, gf, NB_BT, 2 * NB_D, NB_D, 0);

    rope_scale_kernel<<<dim3(NB_BT, NB_H), 128, 0, stream>>>(qf, kf);

    retention_kernel<<<NB_B * NB_H * 8, 256, RET_SMEM, stream>>>(qf, kf, vf, of);

    gated_rmsnorm_kernel<<<dim3(NB_BT, NB_H), 256, 0, stream>>>(of, gf, gnorm_w, ogbf);

    gemm(ogbf, wo, hidden, resid, NB_BT, NB_D, 2 * NB_D, 2);

    rmsnorm_bf16_kernel<<<NB_BT, 256, 0, stream>>>(resid, mlp_norm_w, h2bf, NB_D);

    gemm(h2bf, wgate, nullptr, yf, NB_BT, 2 * NB_INTER, NB_D, 0);

    swiglu_kernel<<<4096, 256, 0, stream>>>(yf, actbf);

    gemm(actbf, wdown, resid, (float*)d_out, NB_BT, NB_D, NB_INTER, 2);
}
